// GCNIIConv_82231443849285
// MI455X (gfx1250) — matrix operand order repaired and checked by reading, not compiled, not run on hardware
//
#include <hip/hip_runtime.h>

typedef __attribute__((ext_vector_type(16))) __bf16 v16bf;
typedef __attribute__((ext_vector_type(8)))  float  v8f;

#define DDIM 128
#define ALPHA_C 0.1f
#define BETA_C 0.40546510810816438f          // log(1.5)
#define OMB_C  0.59453489189183562f          // 1 - log(1.5)

// Native CDNA5 f32 global atomic add (no return -> STOREcnt, resolves in L2).
__device__ __forceinline__ void gatomic_add_f32(float* p, float v) {
  asm volatile("global_atomic_add_f32 %0, %1, off" : : "v"(p), "v"(v) : "memory");
}

// ---------------- kernel 1: h = ALPHA * x0 ----------------
__global__ __launch_bounds__(256) void init_h_kernel(const float4* __restrict__ x0,
                                                     float4* __restrict__ h, int n4) {
  int i = blockIdx.x * 256 + threadIdx.x;
  if (i < n4) {
    float4 v = x0[i];
    float4 r;
    r.x = ALPHA_C * v.x; r.y = ALPHA_C * v.y;
    r.z = ALPHA_C * v.z; r.w = ALPHA_C * v.w;
    h[i] = r;
  }
}

// ---------------- kernel 2: pack B = W^T as bf16 hi/lo in WMMA lane layout ----
// storage index: tid = ((t*4 + c)*32 + L)*16 + e
//   value = B[k][j] = W[j*128 + k],  j = t*16 + (L%16),  k = c*32 + (e/8)*16 + (L/16)*8 + (e%8)
__global__ __launch_bounds__(256) void prep_B_kernel(const float* __restrict__ W,
                                                     __bf16* __restrict__ Bhi,
                                                     __bf16* __restrict__ Blo) {
  int tid = blockIdx.x * 256 + threadIdx.x;     // 0 .. 16383
  int e = tid & 15;
  int L = (tid >> 4) & 31;
  int c = (tid >> 9) & 3;
  int t = tid >> 11;
  int j = t * 16 + (L & 15);
  int k = c * 32 + ((e >> 3) << 4) + ((L >> 4) << 3) + (e & 7);
  float w = W[j * DDIM + k];
  __bf16 hi = (__bf16)w;
  __bf16 lo = (__bf16)(w - (float)hi);
  Bhi[tid] = hi;
  Blo[tid] = lo;
}

// ---------------- kernel 3: scatter-add, one wave per edge ----------------
__global__ __launch_bounds__(256) void scatter_kernel(const float* __restrict__ x,
                                                      const long long* __restrict__ ei,
                                                      const float* __restrict__ nrm,
                                                      float* __restrict__ h, int E) {
  int gid  = blockIdx.x * 256 + threadIdx.x;
  int edge = gid >> 5;
  int lane = gid & 31;
  if (edge >= E) return;
  long long src = ei[edge];                 // edge_index[0][e]
  long long dst = ei[(long long)E + edge];  // edge_index[1][e]
  float s = (1.0f - ALPHA_C) * nrm[edge];
  const float4* xs = (const float4*)(x + src * DDIM);
  float4 v = xs[lane];
  float* hp = h + dst * DDIM + lane * 4;
  gatomic_add_f32(hp + 0, s * v.x);
  gatomic_add_f32(hp + 1, s * v.y);
  gatomic_add_f32(hp + 2, s * v.z);
  gatomic_add_f32(hp + 3, s * v.w);
}

// ---------------- kernel 4: in-place out = (1-b)*h + b*(h @ W^T) ----------------
// one wave32 per 16-row strip; split-precision bf16 WMMA (hi/lo), f32 accumulate
__global__ __launch_bounds__(256) void gemm_residual_kernel(float* __restrict__ h,
                                                            const v16bf* __restrict__ Bhi,
                                                            const v16bf* __restrict__ Blo,
                                                            int ntiles) {
  int wave = blockIdx.x * (256 / 32) + (threadIdx.x >> 5);
  int lane = threadIdx.x & 31;
  if (wave >= ntiles) return;          // uniform per wave: EXEC stays all-ones
  int half = lane >> 4;
  int lm   = lane & 15;
  int m0   = wave * 16;

  // Load A block (rows m0..m0+15, all K) and convert to bf16 hi/lo pairs.
  // lane layout: M = lm, element e -> K = (e/8)*16 + half*8 + (e%8)  (per K-chunk)
  const float* arow = h + (m0 + lm) * DDIM + half * 8;
  v16bf Ahi[4], Alo[4];
#pragma unroll
  for (int c = 0; c < 4; ++c) {
    const float* ap = arow + c * 32;
    float4 q0 = *(const float4*)(ap + 0);
    float4 q1 = *(const float4*)(ap + 4);
    float4 q2 = *(const float4*)(ap + 16);
    float4 q3 = *(const float4*)(ap + 20);
    float f[16];
    f[0]=q0.x; f[1]=q0.y; f[2]=q0.z;  f[3]=q0.w;
    f[4]=q1.x; f[5]=q1.y; f[6]=q1.z;  f[7]=q1.w;
    f[8]=q2.x; f[9]=q2.y; f[10]=q2.z; f[11]=q2.w;
    f[12]=q3.x;f[13]=q3.y;f[14]=q3.z; f[15]=q3.w;
#pragma unroll
    for (int e = 0; e < 16; ++e) {
      __bf16 hi = (__bf16)f[e];
      Ahi[c][e] = hi;
      Alo[c][e] = (__bf16)(f[e] - (float)hi);
    }
  }

#pragma unroll
  for (int t = 0; t < 8; ++t) {
    v8f acc = {0.f, 0.f, 0.f, 0.f, 0.f, 0.f, 0.f, 0.f};
#pragma unroll
    for (int c = 0; c < 4; ++c) {
      v16bf bh = Bhi[(t * 4 + c) * 32 + lane];
      v16bf bl = Blo[(t * 4 + c) * 32 + lane];
      acc = __builtin_amdgcn_wmma_f32_16x16x32_bf16(false, Ahi[c], false, bh, (short)0, acc, false, false);
      acc = __builtin_amdgcn_wmma_f32_16x16x32_bf16(false, Alo[c], false, bh, (short)0, acc, false, false);
      acc = __builtin_amdgcn_wmma_f32_16x16x32_bf16(false, Ahi[c], false, bl, (short)0, acc, false, false);
    }
    // epilogue for this 16x16 tile: C/D layout M = i + 8*half, N = t*16 + lm
#pragma unroll
    for (int i = 0; i < 8; ++i) {
      int row = m0 + i + half * 8;
      int col = t * 16 + lm;
      float hv = h[row * DDIM + col];
      h[row * DDIM + col] = OMB_C * hv + BETA_C * acc[i];
    }
  }
}

extern "C" void kernel_launch(void* const* d_in, const int* in_sizes, int n_in,
                              void* d_out, int out_size, void* d_ws, size_t ws_size,
                              hipStream_t stream) {
  (void)n_in; (void)out_size; (void)ws_size;
  const float*     x   = (const float*)d_in[0];
  const float*     x0  = (const float*)d_in[1];
  const long long* ei  = (const long long*)d_in[2];   // int64 edge_index [2,E]
  const float*     nrm = (const float*)d_in[3];
  const float*     W   = (const float*)d_in[4];
  float* h = (float*)d_out;                           // h built in place in d_out

  int N = in_sizes[0] / DDIM;   // 50000
  int E = in_sizes[3];          // 600000

  __bf16* Bhi = (__bf16*)d_ws;            // 128*128 bf16 = 32 KB
  __bf16* Blo = Bhi + DDIM * DDIM;        // +32 KB

  int n4 = (N * DDIM) / 4;
  init_h_kernel<<<(n4 + 255) / 256, 256, 0, stream>>>((const float4*)x0, (float4*)h, n4);
  prep_B_kernel<<<(DDIM * DDIM) / 256, 256, 0, stream>>>(W, Bhi, Blo);
  scatter_kernel<<<(E * 32 + 255) / 256, 256, 0, stream>>>(x, ei, nrm, h, E);
  int ntiles = N / 16;          // 3125
  gemm_residual_kernel<<<(ntiles + 7) / 8, 256, 0, stream>>>(
      h, (const v16bf*)Bhi, (const v16bf*)Blo, ntiles);
}
